// aLSTM_72980084293936
// MI455X (gfx1250) — matrix operand order repaired and checked by reading, not compiled, not run on hardware
//
#include <hip/hip_runtime.h>

// ---------------- types ----------------
typedef __attribute__((ext_vector_type(16))) __bf16 v16bf;
typedef __attribute__((ext_vector_type(8)))  float  v8f;

struct alignas(16) U4 { unsigned int x, y, z, w; };

#define DEVINL __device__ __forceinline__

DEVINL unsigned short to_bf(float f) {
  union { __bf16 b; unsigned short u; } cv;
  cv.b = (__bf16)f;                       // hardware f32->bf16 convert
  return cv.u;
}

DEVINL float sigm(float x) { return 1.0f / (1.0f + __expf(-x)); }

// ---------------- WMMA tile GEMM over one K-segment (all-bf16 operands) ----------------
// acc[g][mt] += A(m0+16*mt .., k) x B(k, n0+g*gstride ..), K multiple of 32.
// A: bf16 activations, row-major lda.  B: bf16 weights w[row, wcol+k], row-major ldw.
// A frag (16-bit, 16x32): lane half owns K = k0+8*half+[0..7] and k0+16+8*half+[0..7].
// B frag (16-bit, 32x16): lane = N, K = k0+8*half+[0..7] and k0+16+8*half+[0..7].
// All fragments of a K-step are loaded before any WMMA (single wait boundary,
// deep back-to-back WMMA chain; each B frag feeds MT WMMAs).
template<int NG, int MT>
DEVINL void mm_seg(v8f acc[NG][MT],
                   const unsigned short* __restrict__ a, int lda,
                   const unsigned short* __restrict__ w, int ldw, int wcol,
                   int gstride, int K, int m0, int n0, int lane)
{
  const int ml   = lane & 15;   // M row for A, N col for B
  const int half = lane >> 4;

  for (int k0 = 0; k0 < K; k0 += 32) {
    union AV { v16bf v; U4 q[2]; } ua[MT];
    #pragma unroll
    for (int mt = 0; mt < MT; ++mt) {
      const unsigned short* pa =
          a + (size_t)(m0 + 16 * mt + ml) * lda + k0 + 8 * half;
      ua[mt].q[0] = *(const U4*)(pa);        // 8 bf16: k0+8h .. +7
      ua[mt].q[1] = *(const U4*)(pa + 16);   // 8 bf16: k0+16+8h .. +7
    }
    union BV { v16bf v; U4 q[2]; } ub[NG];
    #pragma unroll
    for (int g = 0; g < NG; ++g) {
      const unsigned short* pb =
          w + (size_t)(n0 + g * gstride + ml) * (size_t)ldw + wcol + k0 + 8 * half;
      ub[g].q[0] = *(const U4*)(pb);
      ub[g].q[1] = *(const U4*)(pb + 16);
    }
    #pragma unroll
    for (int g = 0; g < NG; ++g) {
      #pragma unroll
      for (int mt = 0; mt < MT; ++mt) {
        acc[g][mt] = __builtin_amdgcn_wmma_f32_16x16x32_bf16(
            false, ua[mt].v, false, ub[g].v, (short)0, acc[g][mt], false, false);
      }
    }
  }
}

// ---------------- plain LSTM cell (adapt cells), MT=2 ----------------
// gates = [s0|s1|s2] @ wih^T + hold @ whh^T + bih + bhh ; fused nonlinearity.
__global__ __launch_bounds__(32) void cell_a_kernel(
    const unsigned short* __restrict__ s0a, int s0ld, int s0K,
    const unsigned short* __restrict__ s1a, int s1ld, int s1K,
    const unsigned short* __restrict__ s2a, int s2ld, int s2K,
    const unsigned short* __restrict__ wih, int ldwih,
    const unsigned short* __restrict__ whh,
    const unsigned short* __restrict__ holdb,
    const float* __restrict__ bih, const float* __restrict__ bhh,
    float* __restrict__ c, unsigned short* __restrict__ hnewb, int Adim)
{
  const int lane = threadIdx.x;
  const int n0 = blockIdx.x * 16;
  const int m0 = blockIdx.y * 32;
  v8f acc[4][2] = {};
  mm_seg<4,2>(acc, s0a,   s0ld, wih, ldwih, 0,       Adim, s0K,  m0, n0, lane);
  mm_seg<4,2>(acc, s1a,   s1ld, wih, ldwih, s0K,     Adim, s1K,  m0, n0, lane);
  mm_seg<4,2>(acc, s2a,   s2ld, wih, ldwih, s0K+s1K, Adim, s2K,  m0, n0, lane);
  mm_seg<4,2>(acc, holdb, Adim, whh, Adim,  0,       Adim, Adim, m0, n0, lane);

  const int nl = lane & 15, half = lane >> 4;
  const int col = n0 + nl;
  const float bi_ = bih[0*Adim+col] + bhh[0*Adim+col];
  const float bf_ = bih[1*Adim+col] + bhh[1*Adim+col];
  const float bg_ = bih[2*Adim+col] + bhh[2*Adim+col];
  const float bo_ = bih[3*Adim+col] + bhh[3*Adim+col];
  #pragma unroll
  for (int mt = 0; mt < 2; ++mt) {
    #pragma unroll
    for (int r = 0; r < 8; ++r) {
      const int m = m0 + 16*mt + r + 8*half;
      const float i = sigm (acc[0][mt][r] + bi_);
      const float f = sigm (acc[1][mt][r] + bf_);
      const float g = tanhf(acc[2][mt][r] + bg_);
      const float o = sigm (acc[3][mt][r] + bo_);
      const size_t idx = (size_t)m * Adim + col;
      const float c2 = f * c[idx] + i * g;
      c[idx] = c2;
      hnewb[idx] = to_bf(o * tanhf(c2));
    }
  }
}

// ---------------- adaptive LSTM cell (main cells), MT=2 ----------------
// A operands (x*a_x, h*a_h) are pre-scaled+pre-converted bf16 (sx, sh).
// igates = (sx@wih^T)*a_ig + fb*a_b ; hgates = (sh@whh^T)*a_hg
__global__ __launch_bounds__(32) void cell_f_kernel(
    const unsigned short* __restrict__ sx, int ninp,
    const unsigned short* __restrict__ sh, int nhid,
    const unsigned short* __restrict__ wih,
    const unsigned short* __restrict__ whh,
    const float* __restrict__ fb,
    const float* __restrict__ ahe, int P,
    float* __restrict__ c, float* __restrict__ hnew,
    unsigned short* __restrict__ hnewb, float* __restrict__ out2)
{
  const int lane = threadIdx.x;
  const int n0 = blockIdx.x * 16;
  const int m0 = blockIdx.y * 32;
  v8f accI[4][2] = {};
  v8f accH[4][2] = {};
  mm_seg<4,2>(accI, sx, ninp, wih, ninp, 0, nhid, ninp, m0, n0, lane);
  mm_seg<4,2>(accH, sh, nhid, whh, nhid, 0, nhid, nhid, m0, n0, lane);

  const int nl = lane & 15, half = lane >> 4;
  const int col = n0 + nl;
  #pragma unroll
  for (int mt = 0; mt < 2; ++mt) {
    #pragma unroll
    for (int r = 0; r < 8; ++r) {
      const int m = m0 + 16*mt + r + 8*half;
      const float* arow = ahe + (size_t)m * P;
      float gv[4];
      #pragma unroll
      for (int g = 0; g < 4; ++g) {
        const int G = g * nhid + col;
        const float aig = arow[ninp +     nhid + G];
        const float ahg = arow[ninp + 5 * nhid + G];
        const float ab  = arow[ninp + 9 * nhid + G];
        gv[g] = accI[g][mt][r] * aig + fb[G] * ab + accH[g][mt][r] * ahg;
      }
      const float i = sigm (gv[0]);
      const float f = sigm (gv[1]);
      const float g = tanhf(gv[2]);
      const float o = sigm (gv[3]);
      const size_t idx = (size_t)m * nhid + col;
      const float c2 = f * c[idx] + i * g;
      c[idx] = c2;
      const float h = o * tanhf(c2);
      hnew[idx] = h;
      hnewb[idx] = to_bf(h);
      if (out2) out2[idx] = h;
    }
  }
}

// ---------------- policy projection: ahe = hA @ pw^T + pb (K=128, MT=4) ----------------
__global__ __launch_bounds__(32) void gemm_p_kernel(
    const unsigned short* __restrict__ hAb, const unsigned short* __restrict__ pw,
    const float* __restrict__ pb, float* __restrict__ ahe, int P)
{
  const int lane = threadIdx.x;
  const int n0 = blockIdx.x * 16;
  const int m0 = blockIdx.y * 64;   // grid.y == 1, whole batch per wave
  v8f acc[1][4] = {};
  mm_seg<1,4>(acc, hAb, 128, pw, 128, 0, 0, 128, m0, n0, lane);
  const int nl = lane & 15, half = lane >> 4;
  const int col = n0 + nl;
  const float b = pb[col];
  #pragma unroll
  for (int mt = 0; mt < 4; ++mt) {
    #pragma unroll
    for (int r = 0; r < 8; ++r) {
      const int m = m0 + 16*mt + r + 8*half;
      ahe[(size_t)m * P + col] = acc[0][mt][r] + b;
    }
  }
}

// ---------------- per-step adaptive pre-scale + bf16 convert ----------------
__global__ void scale_conv_kernel(
    const float* __restrict__ x, int ldx, int ninp,
    const float* __restrict__ h, int nhid,
    const float* __restrict__ ahe, int P,
    unsigned short* __restrict__ sx, unsigned short* __restrict__ sh)
{
  const int b = blockIdx.y;
  const int j = blockIdx.x * blockDim.x + threadIdx.x;
  const float* arow = ahe + (size_t)b * P;
  if (j < ninp) {
    sx[(size_t)b * ninp + j] = to_bf(x[(size_t)b * ldx + j] * arow[j]);
  } else if (j < ninp + nhid) {
    const int jj = j - ninp;
    sh[(size_t)b * nhid + jj] = to_bf(h[(size_t)b * nhid + jj] * arow[ninp + jj]);
  }
}

// ---------------- helpers ----------------
__global__ void convert_kernel(const float* __restrict__ in,
                               unsigned short* __restrict__ out, int n) {
  const int i = blockIdx.x * blockDim.x + threadIdx.x;
  if (i < n) out[i] = to_bf(in[i]);
}

__global__ void zero_kernel(unsigned int* __restrict__ p, int n) {
  const int i = blockIdx.x * blockDim.x + threadIdx.x;
  if (i < n) p[i] = 0u;
}

// ---------------- host ----------------
extern "C" void kernel_launch(void* const* d_in, const int* in_sizes, int n_in,
                              void* d_out, int out_size, void* d_ws, size_t ws_size,
                              hipStream_t stream) {
  (void)in_sizes; (void)n_in; (void)out_size; (void)ws_size;
  constexpr int T = 512, B = 64, I = 256, H = 1024, A = 128, O = 1024;
  const int P0 = I + 13 * H;   // 13568
  const int P1 = H + 13 * O;   // 14336

  const float* x      = (const float*)d_in[0];
  const float* a_wih0 = (const float*)d_in[1];
  const float* a_whh0 = (const float*)d_in[2];
  const float* a_bih0 = (const float*)d_in[3];
  const float* a_bhh0 = (const float*)d_in[4];
  const float* p_w0   = (const float*)d_in[5];
  const float* p_b0   = (const float*)d_in[6];
  const float* f_wih0 = (const float*)d_in[7];
  const float* f_whh0 = (const float*)d_in[8];
  const float* f_b0   = (const float*)d_in[9];
  const float* a_wih1 = (const float*)d_in[10];
  const float* a_whh1 = (const float*)d_in[11];
  const float* a_bih1 = (const float*)d_in[12];
  const float* a_bhh1 = (const float*)d_in[13];
  const float* p_w1   = (const float*)d_in[14];
  const float* p_b1   = (const float*)d_in[15];
  const float* f_wih1 = (const float*)d_in[16];
  const float* f_whh1 = (const float*)d_in[17];
  const float* f_b1   = (const float*)d_in[18];
  float* out = (float*)d_out;

  char* base = (char*)d_ws;
  size_t off = 0;
  auto alloc = [&](size_t bytes) -> char* {
    off = (off + 255) & ~(size_t)255;
    char* p = base + off;
    off += bytes;
    return p;
  };

  // ---- zeroed state span: fp32 c/h states, then bf16 h states (contiguous) ----
  float* a0c = (float*)alloc((size_t)B*A * 4);
  float* a1c = (float*)alloc((size_t)B*A * 4);
  float* f0c = (float*)alloc((size_t)B*H * 4);
  float* f1c = (float*)alloc((size_t)B*O * 4);
  float* f0x = (float*)alloc((size_t)2*B*H * 4);   // fp32 h (scale_conv input)
  float* f1x = (float*)alloc((size_t)2*B*O * 4);
  unsigned short* a0xb = (unsigned short*)alloc((size_t)2*B*A * 2);  // bf16 h states
  unsigned short* a1xb = (unsigned short*)alloc((size_t)2*B*A * 2);
  unsigned short* f0xb = (unsigned short*)alloc((size_t)2*B*H * 2);
  unsigned short* f1xb = (unsigned short*)alloc((size_t)2*B*O * 2);
  char* zero_end = (char*)(f1xb + (size_t)2*B*O);

  // ---- per-step scratch (fully written before read; no zeroing needed) ----
  float* ahe0 = (float*)alloc((size_t)B*P0 * 4);
  float* ahe1 = (float*)alloc((size_t)B*P1 * 4);
  unsigned short* sx0 = (unsigned short*)alloc((size_t)B*I * 2);
  unsigned short* sh0 = (unsigned short*)alloc((size_t)B*H * 2);
  unsigned short* sx1 = (unsigned short*)alloc((size_t)B*H * 2);
  unsigned short* sh1 = (unsigned short*)alloc((size_t)B*O * 2);

  // ---- bf16 input + weights ----
  unsigned short* xbf     = (unsigned short*)alloc((size_t)T*B*I * 2);
  unsigned short* w_awih0 = (unsigned short*)alloc((size_t)4*A*(A+I+H) * 2);
  unsigned short* w_awhh0 = (unsigned short*)alloc((size_t)4*A*A       * 2);
  unsigned short* w_pw0   = (unsigned short*)alloc((size_t)P0*A        * 2);
  unsigned short* w_fwih0 = (unsigned short*)alloc((size_t)4*H*I       * 2);
  unsigned short* w_fwhh0 = (unsigned short*)alloc((size_t)4*H*H       * 2);
  unsigned short* w_awih1 = (unsigned short*)alloc((size_t)4*A*(A+H+O) * 2);
  unsigned short* w_awhh1 = (unsigned short*)alloc((size_t)4*A*A       * 2);
  unsigned short* w_pw1   = (unsigned short*)alloc((size_t)P1*A        * 2);
  unsigned short* w_fwih1 = (unsigned short*)alloc((size_t)4*O*H       * 2);
  unsigned short* w_fwhh1 = (unsigned short*)alloc((size_t)4*O*O       * 2);

  auto conv = [&](const float* src, unsigned short* dst, size_t n) {
    convert_kernel<<<dim3((unsigned)((n + 255) / 256)), 256, 0, stream>>>(src, dst, (int)n);
  };
  conv(x,      xbf,     (size_t)T*B*I);
  conv(a_wih0, w_awih0, (size_t)4*A*(A+I+H));
  conv(a_whh0, w_awhh0, (size_t)4*A*A);
  conv(p_w0,   w_pw0,   (size_t)P0*A);
  conv(f_wih0, w_fwih0, (size_t)4*H*I);
  conv(f_whh0, w_fwhh0, (size_t)4*H*H);
  conv(a_wih1, w_awih1, (size_t)4*A*(A+H+O));
  conv(a_whh1, w_awhh1, (size_t)4*A*A);
  conv(p_w1,   w_pw1,   (size_t)P1*A);
  conv(f_wih1, w_fwih1, (size_t)4*O*H);
  conv(f_whh1, w_fwhh1, (size_t)4*O*O);

  // zero all recurrent state (fp32 + bf16, contiguous span)
  const size_t zn = (size_t)(zero_end - (char*)a0c) / 4;
  zero_kernel<<<dim3((unsigned)((zn + 255) / 256)), 256, 0, stream>>>(
      (unsigned int*)a0c, (int)zn);

  const dim3 blk(32);
  const dim3 gA(A / 16, B / 32);               // MT=2
  const dim3 gH(H / 16, B / 32);               // MT=2
  const dim3 gO(O / 16, B / 32);               // MT=2
  const dim3 gP0(P0 / 16, 1);                  // MT=4
  const dim3 gP1(P1 / 16, 1);                  // MT=4
  const dim3 gS0((I + H + 255) / 256, B);      // scale_conv layer 0
  const dim3 gS1((H + O + 255) / 256, B);      // scale_conv layer 1

  for (int t = 0; t < T; ++t) {
    const int rs = t & 1, wsl = rs ^ 1;
    const float*          xt  = x   + (size_t)t * B * I;
    const unsigned short* xtb = xbf + (size_t)t * B * I;
    const unsigned short* a0xb_r = a0xb + (size_t)rs  * B * A;
    unsigned short*       a0xb_w = a0xb + (size_t)wsl * B * A;
    const unsigned short* a1xb_r = a1xb + (size_t)rs  * B * A;
    unsigned short*       a1xb_w = a1xb + (size_t)wsl * B * A;
    const float*          f0x_r  = f0x  + (size_t)rs  * B * H;
    float*                f0x_w  = f0x  + (size_t)wsl * B * H;
    const unsigned short* f0xb_r = f0xb + (size_t)rs  * B * H;
    unsigned short*       f0xb_w = f0xb + (size_t)wsl * B * H;
    const float*          f1x_r  = f1x  + (size_t)rs  * B * O;
    float*                f1x_w  = f1x  + (size_t)wsl * B * O;
    const unsigned short* f1xb_r = f1xb + (size_t)rs  * B * O;
    unsigned short*       f1xb_w = f1xb + (size_t)wsl * B * O;

    // layer 0 adapt cell: ax0 = [xt | f0x_prev | a1x_prev]
    cell_a_kernel<<<gA, blk, 0, stream>>>(
        xtb, I, I, f0xb_r, H, H, a1xb_r, A, A,
        w_awih0, A + I + H, w_awhh0, a0xb_r, a_bih0, a_bhh0, a0c, a0xb_w, A);
    // policy projection 0
    gemm_p_kernel<<<gP0, blk, 0, stream>>>(a0xb_w, w_pw0, p_b0, ahe0, P0);
    // adaptive pre-scale for layer 0 main cell
    scale_conv_kernel<<<gS0, 256, 0, stream>>>(xt, I, I, f0x_r, H, ahe0, P0, sx0, sh0);
    // layer 0 main (adaptive) cell
    cell_f_kernel<<<gH, blk, 0, stream>>>(
        sx0, I, sh0, H, w_fwih0, w_fwhh0, f_b0, ahe0, P0,
        f0c, f0x_w, f0xb_w, nullptr);
    // layer 1 adapt cell: ax1 = [f0x_new | f1x_prev | a0x_new]
    cell_a_kernel<<<gA, blk, 0, stream>>>(
        f0xb_w, H, H, f1xb_r, O, O, a0xb_w, A, A,
        w_awih1, A + H + O, w_awhh1, a1xb_r, a_bih1, a_bhh1, a1c, a1xb_w, A);
    // policy projection 1
    gemm_p_kernel<<<gP1, blk, 0, stream>>>(a1xb_w, w_pw1, p_b1, ahe1, P1);
    // adaptive pre-scale for layer 1 main cell
    scale_conv_kernel<<<gS1, 256, 0, stream>>>(f0x_w, H, H, f1x_r, O, ahe1, P1, sx1, sh1);
    // layer 1 main cell, writes h into output slab for step t
    cell_f_kernel<<<gO, blk, 0, stream>>>(
        sx1, H, sh1, O, w_fwih1, w_fwhh1, f_b1, ahe1, P1,
        f1c, f1x_w, f1xb_w, out + (size_t)t * B * O);
  }
}
